// GALSTM_30975304139537
// MI455X (gfx1250) — compile-verified, matrix operand order checked by reading, not run on hardware
//
#include <hip/hip_runtime.h>
#include <stdint.h>

// ---------- types ----------
typedef __attribute__((ext_vector_type(16))) __bf16 v16bf;
typedef __attribute__((ext_vector_type(8)))  float  v8f;
typedef __attribute__((ext_vector_type(4)))  unsigned int v4u;

#define CCH 64      // channels
#define HDS 2       // heads
#define NCONV 6

// ---------- small device helpers ----------
__device__ __forceinline__ float sigmoidf_(float x) { return 1.0f / (1.0f + __expf(-x)); }

__device__ __forceinline__ unsigned encf(float f) {
    unsigned u = __float_as_uint(f);
    return (u & 0x80000000u) ? ~u : (u | 0x80000000u);   // monotone float->uint
}
__device__ __forceinline__ float decf(unsigned u) {
    u = (u & 0x80000000u) ? (u & 0x7FFFFFFFu) : ~u;
    return __uint_as_float(u);
}
__device__ __forceinline__ void atomAddF(float* p, float v) {
    __hip_atomic_fetch_add(p, v, __ATOMIC_RELAXED, __HIP_MEMORY_SCOPE_AGENT);
}

// ---------- K1: mean over edges of tanh(interval*W_ti+b_ti), tanh(span*W_ts+b_ts) ----------
__global__ void __launch_bounds__(256)
k_timefeat_reduce(const float* __restrict__ intervals, const float* __restrict__ spans,
                  const float* __restrict__ W_ti, const float* __restrict__ b_ti,
                  const float* __restrict__ W_ts, const float* __restrict__ b_ts,
                  float* __restrict__ tif_sum, float* __restrict__ tsf_sum, int E) {
    int c = threadIdx.x & 63;
    int q = threadIdx.x >> 6;                       // 0..3
    int chunk  = blockIdx.x * 4 + q;
    int nchunk = gridDim.x * 4;
    float wi = W_ti[c], bi = b_ti[c];
    float ws = W_ts[c], bs = b_ts[c];
    float pi = 0.f, ps = 0.f;
    for (int e = chunk; e < E; e += nchunk) {
        pi += tanhf(intervals[e] * wi + bi);
        ps += tanhf(spans[e]     * ws + bs);
    }
    atomAddF(tif_sum + c, pi);
    atomAddF(tsf_sum + c, ps);
}

// ---------- K1b: small matvecs (1 block) ----------
// tiv1 = tif_mean @ W_ti1 + b_ti1 ; tiv2 = tif_mean @ W_ti2 ; same for ts.
// was[k][f][h] = sum_c Wg[k][f][h][c]*att_src[k][h][c] ; wad with att_dst.
__global__ void __launch_bounds__(256)
k_small_matvec(const float* __restrict__ tif_sum, const float* __restrict__ tsf_sum,
               const float* __restrict__ W_ti1, const float* __restrict__ b_ti1,
               const float* __restrict__ W_ti2,
               const float* __restrict__ W_ts1, const float* __restrict__ b_ts1,
               const float* __restrict__ W_ts2,
               const float* __restrict__ Wg, const float* __restrict__ att_src,
               const float* __restrict__ att_dst,
               float* __restrict__ tif_mean, float* __restrict__ tsf_mean,
               float* __restrict__ tiv1, float* __restrict__ tiv2,
               float* __restrict__ tsv1, float* __restrict__ tsv2,
               float* __restrict__ was, float* __restrict__ wad, int E) {
    int t = threadIdx.x;
    if (t < 64) {
        tif_mean[t] = tif_sum[t] / (float)E;
        tsf_mean[t] = tsf_sum[t] / (float)E;
    }
    __threadfence_block();
    __syncthreads();
    if (t < 64) {
        float a1 = b_ti1[t], a2 = 0.f, s1 = b_ts1[t], s2 = 0.f;
        for (int f = 0; f < 64; ++f) {
            float mi = tif_mean[f], ms = tsf_mean[f];
            a1 += mi * W_ti1[f * 64 + t];
            a2 += mi * W_ti2[f * 64 + t];
            s1 += ms * W_ts1[f * 64 + t];
            s2 += ms * W_ts2[f * 64 + t];
        }
        tiv1[t] = a1; tiv2[t] = a2; tsv1[t] = s1; tsv2[t] = s2;
    }
    for (int i = t; i < NCONV * 64 * HDS; i += 256) {
        int k = i >> 7, r = i & 127, f = r >> 1, h = r & 1;
        const float* wrow = Wg + (((size_t)k * 64 + f) * HDS + h) * 64;
        const float* as   = att_src + ((size_t)k * HDS + h) * 64;
        const float* ad   = att_dst + ((size_t)k * HDS + h) * 64;
        float ss = 0.f, sd = 0.f;
        for (int c = 0; c < 64; ++c) { ss += wrow[c] * as[c]; sd += wrow[c] * ad[c]; }
        was[i] = ss; wad[i] = sd;
    }
}

// ---------- K2a: X f32 -> bf16 ----------
__global__ void __launch_bounds__(256)
k_cvt_x(const float* __restrict__ X, __bf16* __restrict__ Xb, int n) {
    int i = blockIdx.x * 256 + threadIdx.x;
    if (i < n) Xb[i] = (__bf16)X[i];
}

// ---------- K2b: pack Wg into WMMA-B fragment order (bf16) ----------
// fragment id = ((k*8+t)*2+s)*32 + lane, 16 contiguous halves each.
// element j of lane: b[j] = W[k][ K(j,hi)+s*32 ][ t*16 + (lane&15) ],  K(j,hi) = j<8 ? j+hi*8 : 8+j+hi*8
__global__ void __launch_bounds__(256)
k_pack_w(const float* __restrict__ Wg, __bf16* __restrict__ Wfrag) {
    int id = blockIdx.x * 256 + threadIdx.x;
    if (id >= NCONV * 8 * 2 * 32) return;
    int lane = id & 31, s = (id >> 5) & 1, t = (id >> 6) & 7, k = id >> 9;
    int hi = lane >> 4, n = lane & 15;
    __bf16* dst = Wfrag + (size_t)id * 16;
    for (int j = 0; j < 16; ++j) {
        int kk  = s * 32 + hi * 8 + ((j < 8) ? j : 8 + j);
        int col = t * 16 + n;                           // col = h*64 + c over [128]
        dst[j] = (__bf16)Wg[((size_t)k * 64 + kk) * 128 + col];
    }
}

// ---------- K3: projection h[k] = X @ W[k] via v_wmma_f32_16x16x32_bf16 ----------
// one wave -> 16 rows x 128 cols (8 accumulators, 16 WMMAs); 8 waves/block -> 128 rows.
__global__ void __launch_bounds__(256)
k_proj_wmma(const __bf16* __restrict__ Xb, const __bf16* __restrict__ Wfrag,
            __bf16* __restrict__ hout, int N) {
    int wave = threadIdx.x >> 5;
    int lane = threadIdx.x & 31;
    int k = blockIdx.y;
    int rowbase = blockIdx.x * 128 + wave * 16;
    if (rowbase >= N) return;
    int m = lane & 15, hi = lane >> 4;
    const __bf16* xrow = Xb + (size_t)(rowbase + m) * 64;

    v8f acc[8];
    for (int t = 0; t < 8; ++t) acc[t] = (v8f){0,0,0,0,0,0,0,0};

    for (int s = 0; s < 2; ++s) {
        union { v16bf v; v4u q[2]; } ua;
        ua.q[0] = *(const v4u*)(xrow + s * 32 + hi * 8);         // K = s*32+hi*8 .. +7
        ua.q[1] = *(const v4u*)(xrow + s * 32 + 16 + hi * 8);    // K = s*32+16+hi*8 .. +7
        for (int t = 0; t < 8; ++t) {
            union { v16bf v; v4u q[2]; } ub;
            const __bf16* bp = Wfrag + ((((size_t)k * 8 + t) * 2 + s) * 32 + lane) * 16;
            ub.q[0] = ((const v4u*)bp)[0];
            ub.q[1] = ((const v4u*)bp)[1];
            acc[t] = __builtin_amdgcn_wmma_f32_16x16x32_bf16(
                false, ua.v, false, ub.v, (short)0, acc[t], false, false);
        }
    }
    // D layout: VGPR r -> row = rowbase + hi*8 + r ; col = t*16 + (lane&15)
    int nn = lane & 15;
    for (int t = 0; t < 8; ++t) {
        int col = t * 16 + nn;
        for (int r = 0; r < 8; ++r) {
            int row = rowbase + hi * 8 + r;
            hout[((size_t)k * N + row) * 128 + col] = (__bf16)acc[t][r];
        }
    }
}

// ---------- K4: es/ed = X @ was / wad  (attention logit halves per node) ----------
__global__ void __launch_bounds__(64)
k_esed(const float* __restrict__ X, const float* __restrict__ was, const float* __restrict__ wad,
       float* __restrict__ es, float* __restrict__ ed, int N) {
    __shared__ float Xs[64][65];
    int nb = blockIdx.x * 64;
    int t = threadIdx.x;
    for (int j = 0; j < 64; ++j) {
        int row = nb + j;
        Xs[j][t] = (row < N) ? X[(size_t)row * 64 + t] : 0.f;
    }
    __syncthreads();
    int row = nb + t;
    if (row >= N) return;
    for (int k = 0; k < NCONV; ++k) {
        float a0 = 0.f, a1 = 0.f, d0 = 0.f, d1 = 0.f;
        const float* ws = was + k * 128;
        const float* wd = wad + k * 128;
        for (int f = 0; f < 64; ++f) {
            float x = Xs[t][f];
            a0 += x * ws[f * 2 + 0]; a1 += x * ws[f * 2 + 1];
            d0 += x * wd[f * 2 + 0]; d1 += x * wd[f * 2 + 1];
        }
        size_t o = ((size_t)k * N + row) * 2;
        es[o] = a0; es[o + 1] = a1;
        ed[o] = d0; ed[o + 1] = d1;
    }
}

// ---------- K5: logits + segment max (encoded uint atomicMax) ----------
__global__ void __launch_bounds__(256)
k_attn_logits(const int* __restrict__ srcA, const int* __restrict__ dstA,
              const float* __restrict__ es, const float* __restrict__ ed,
              float* __restrict__ eebuf, unsigned* __restrict__ emax, int N, int E) {
    int ET = E + N;
    int e = blockIdx.x * 256 + threadIdx.x;
    if (e >= ET) return;
    int s = (e < E) ? srcA[e] : (e - E);
    int d = (e < E) ? dstA[e] : (e - E);
    for (int k = 0; k < NCONV; ++k) {
        float2 a = *(const float2*)(es + ((size_t)k * N + s) * 2);
        float2 b = *(const float2*)(ed + ((size_t)k * N + d) * 2);
        float v0 = a.x + b.x; v0 = (v0 > 0.f) ? v0 : 0.2f * v0;   // leaky_relu 0.2
        float v1 = a.y + b.y; v1 = (v1 > 0.f) ? v1 : 0.2f * v1;
        float2 o; o.x = v0; o.y = v1;
        *(float2*)(eebuf + ((size_t)k * ET + e) * 2) = o;
        atomicMax(emax + ((size_t)k * N + d) * 2 + 0, encf(v0));
        atomicMax(emax + ((size_t)k * N + d) * 2 + 1, encf(v1));
    }
}

// ---------- K6: exp(e - max) + segment sum ----------
__global__ void __launch_bounds__(256)
k_attn_expsum(const int* __restrict__ dstA, const unsigned* __restrict__ emax,
              float* __restrict__ eebuf, float* __restrict__ den, int N, int E) {
    int ET = E + N;
    int e = blockIdx.x * 256 + threadIdx.x;
    if (e >= ET) return;
    int d = (e < E) ? dstA[e] : (e - E);
    for (int k = 0; k < NCONV; ++k) {
        float2 v = *(const float2*)(eebuf + ((size_t)k * ET + e) * 2);
        float m0 = decf(emax[((size_t)k * N + d) * 2 + 0]);
        float m1 = decf(emax[((size_t)k * N + d) * 2 + 1]);
        float e0 = __expf(v.x - m0), e1 = __expf(v.y - m1);
        float2 o; o.x = e0; o.y = e1;
        *(float2*)(eebuf + ((size_t)k * ET + e) * 2) = o;
        atomAddF(den + ((size_t)k * N + d) * 2 + 0, e0);
        atomAddF(den + ((size_t)k * N + d) * 2 + 1, e1);
    }
}

// ---------- K7: alpha-weighted scatter, fused head-mean (1 wave per (k,edge)) ----------
__global__ void __launch_bounds__(256)
k_scatter(const int* __restrict__ srcA, const int* __restrict__ dstA,
          const float* __restrict__ eebuf, const float* __restrict__ den,
          const __bf16* __restrict__ hbuf, float* __restrict__ outm, int N, int E) {
    int ET = E + N;
    long total = (long)NCONV * ET;
    int lane = threadIdx.x & 31;
    long wid = (long)blockIdx.x * 8 + (threadIdx.x >> 5);
    long nw  = (long)gridDim.x * 8;
    for (long task = wid; task < total; task += nw) {
        int k = (int)(task / ET);
        int e = (int)(task - (long)k * ET);
        int s = (e < E) ? srcA[e] : (e - E);
        int d = (e < E) ? dstA[e] : (e - E);
        float2 ee = *(const float2*)(eebuf + ((size_t)k * ET + e) * 2);
        float2 dn = *(const float2*)(den   + ((size_t)k * N  + d) * 2);
        float a0 = ee.x / (dn.x + 1e-16f);
        float a1 = ee.y / (dn.y + 1e-16f);
        const unsigned* hp = (const unsigned*)(hbuf + ((size_t)k * N + s) * 128);
        unsigned u0 = hp[lane];        // head0: channels 2*lane, 2*lane+1 (bf16 pair)
        unsigned u1 = hp[32 + lane];   // head1: channels 2*lane, 2*lane+1
        float h00 = __uint_as_float((u0 & 0xFFFFu) << 16);
        float h01 = __uint_as_float(u0 & 0xFFFF0000u);
        float h10 = __uint_as_float((u1 & 0xFFFFu) << 16);
        float h11 = __uint_as_float(u1 & 0xFFFF0000u);
        float* op = outm + ((size_t)k * N + d) * 64 + lane * 2;
        atomAddF(op,     0.5f * (a0 * h00 + a1 * h10));
        atomAddF(op + 1, 0.5f * (a0 * h01 + a1 * h11));
    }
}

// ---------- K8: LSTM gates ----------
__global__ void __launch_bounds__(256)
k_gates(const float* __restrict__ outm, const float* __restrict__ Cin,
        const float* __restrict__ bg,
        const float* __restrict__ tiv1, const float* __restrict__ tsv1,
        const float* __restrict__ tiv2, const float* __restrict__ tsv2,
        const float* __restrict__ b_i, const float* __restrict__ b_f,
        const float* __restrict__ b_c, const float* __restrict__ b_o,
        float* __restrict__ out, int N) {
    size_t NC = (size_t)N * 64;
    size_t i = (size_t)blockIdx.x * 256 + threadIdx.x;
    if (i >= NC) return;
    int c = (int)(i & 63);
    float g0 = outm[0 * NC + i] + bg[0 * 64 + c];
    float g1 = outm[1 * NC + i] + bg[1 * 64 + c];
    float g2 = outm[2 * NC + i] + bg[2 * 64 + c];
    float g3 = outm[3 * NC + i] + bg[3 * 64 + c];
    float g4 = outm[4 * NC + i] + bg[4 * 64 + c];
    float g5 = outm[5 * NC + i] + bg[5 * 64 + c];
    float ti = sigmoidf_(g0 + tiv1[c]);
    float si = sigmoidf_(g1 + tsv1[c]);
    float Ig = sigmoidf_(g2 + b_i[c]);
    float Fg = sigmoidf_(g3 + b_f[c]);
    float Tg = tanhf(g4 + b_c[c]);
    float Cn = Fg * ti * Cin[i] + Ig * si * Tg;
    float Og = sigmoidf_(g5 + tiv2[c] + tsv2[c] + b_o[c]);
    out[i]      = Og * tanhf(Cn);   // H_new
    out[NC + i] = Cn;               // C_new
}

// ---------- launch ----------
extern "C" void kernel_launch(void* const* d_in, const int* in_sizes, int n_in,
                              void* d_out, int out_size, void* d_ws, size_t ws_size,
                              hipStream_t stream) {
    const float* X         = (const float*)d_in[0];
    const int*   ei        = (const int*)  d_in[1];
    const float* intervals = (const float*)d_in[2];
    const float* spans     = (const float*)d_in[3];
    const float* Cin       = (const float*)d_in[5];
    const float* Wg        = (const float*)d_in[6];
    const float* att_src   = (const float*)d_in[7];
    const float* att_dst   = (const float*)d_in[8];
    const float* bg        = (const float*)d_in[9];
    const float* W_ti      = (const float*)d_in[10];
    const float* b_ti      = (const float*)d_in[11];
    const float* W_ti1     = (const float*)d_in[12];
    const float* b_ti1     = (const float*)d_in[13];
    const float* W_ti2     = (const float*)d_in[14];
    const float* W_ts      = (const float*)d_in[15];
    const float* b_ts      = (const float*)d_in[16];
    const float* W_ts1     = (const float*)d_in[17];
    const float* b_ts1     = (const float*)d_in[18];
    const float* W_ts2     = (const float*)d_in[19];
    const float* b_i       = (const float*)d_in[20];
    const float* b_f       = (const float*)d_in[21];
    const float* b_c       = (const float*)d_in[22];
    const float* b_o       = (const float*)d_in[23];

    const int N  = in_sizes[0] / 64;
    const int E  = in_sizes[1] / 2;
    const int ET = E + N;
    const int* srcA = ei;
    const int* dstA = ei + E;

    // workspace carve-up (256B aligned regions)
    char* wp = (char*)d_ws;
    auto carve = [&](size_t bytes) -> char* {
        char* p = wp; wp += (bytes + 255) & ~(size_t)255; return p;
    };
    __bf16*   Xb     = (__bf16*)  carve((size_t)N * 64 * 2);
    __bf16*   Wfrag  = (__bf16*)  carve((size_t)NCONV * 8 * 2 * 32 * 16 * 2);
    __bf16*   hbuf   = (__bf16*)  carve((size_t)NCONV * N * 128 * 2);
    float*    es     = (float*)   carve((size_t)NCONV * N * 2 * 4);
    float*    ed     = (float*)   carve((size_t)NCONV * N * 2 * 4);
    unsigned* emax   = (unsigned*)carve((size_t)NCONV * N * 2 * 4);
    float*    den    = (float*)   carve((size_t)NCONV * N * 2 * 4);
    float*    eebuf  = (float*)   carve((size_t)NCONV * ET * 2 * 4);
    float*    outm   = (float*)   carve((size_t)NCONV * N * 64 * 4);
    float*    tif_sum  = (float*) carve(64 * 4);
    float*    tsf_sum  = (float*) carve(64 * 4);
    float*    tif_mean = (float*) carve(64 * 4);
    float*    tsf_mean = (float*) carve(64 * 4);
    float*    tiv1     = (float*) carve(64 * 4);
    float*    tiv2     = (float*) carve(64 * 4);
    float*    tsv1     = (float*) carve(64 * 4);
    float*    tsv2     = (float*) carve(64 * 4);
    float*    was      = (float*) carve((size_t)NCONV * 64 * 2 * 4);
    float*    wad      = (float*) carve((size_t)NCONV * 64 * 2 * 4);

    // zero accumulators (graph-capture-safe memset nodes)
    hipMemsetAsync(emax,    0, (size_t)NCONV * N * 2 * 4, stream);
    hipMemsetAsync(den,     0, (size_t)NCONV * N * 2 * 4, stream);
    hipMemsetAsync(outm,    0, (size_t)NCONV * N * 64 * 4, stream);
    hipMemsetAsync(tif_sum, 0, 64 * 4, stream);
    hipMemsetAsync(tsf_sum, 0, 64 * 4, stream);

    // time-feature means + small matvecs
    k_timefeat_reduce<<<512, 256, 0, stream>>>(intervals, spans, W_ti, b_ti, W_ts, b_ts,
                                               tif_sum, tsf_sum, E);
    k_small_matvec<<<1, 256, 0, stream>>>(tif_sum, tsf_sum, W_ti1, b_ti1, W_ti2,
                                          W_ts1, b_ts1, W_ts2, Wg, att_src, att_dst,
                                          tif_mean, tsf_mean, tiv1, tiv2, tsv1, tsv2,
                                          was, wad, E);

    // bf16 conversion + W fragment packing
    k_cvt_x<<<(N * 64 + 255) / 256, 256, 0, stream>>>(X, Xb, N * 64);
    k_pack_w<<<(NCONV * 8 * 2 * 32 + 255) / 256, 256, 0, stream>>>(Wg, Wfrag);

    // WMMA projections: 6 x [N,64] @ [64,128]
    dim3 gproj((N + 127) / 128, NCONV);
    k_proj_wmma<<<gproj, 256, 0, stream>>>(Xb, Wfrag, hbuf, N);

    // attention logit halves per node
    k_esed<<<(N + 63) / 64, 64, 0, stream>>>(X, was, wad, es, ed, N);

    // edge softmax passes
    k_attn_logits<<<(ET + 255) / 256, 256, 0, stream>>>(srcA, dstA, es, ed, eebuf, emax, N, E);
    k_attn_expsum<<<(ET + 255) / 256, 256, 0, stream>>>(dstA, emax, eebuf, den, N, E);

    // weighted scatter (fused head mean)
    k_scatter<<<4096, 256, 0, stream>>>(srcA, dstA, eebuf, den, hbuf, outm, N, E);

    // LSTM gates -> (H_new, C_new)
    k_gates<<<(int)(((size_t)N * 64 + 255) / 256), 256, 0, stream>>>(
        outm, Cin, bg, tiv1, tsv1, tiv2, tsv2, b_i, b_f, b_c, b_o, (float*)d_out, N);
}
